// Model_DropEdge_87033217286854
// MI455X (gfx1250) — matrix operand order repaired and checked by reading, not compiled, not run on hardware
//
#include <hip/hip_runtime.h>
#include <hip/hip_bf16.h>
#include <stdint.h>

// GCN dims (fixed by the reference)
#define FEAT 256
#define HID  128
#define OUTF 64

typedef __attribute__((ext_vector_type(16))) __bf16 v16bf;
typedef __attribute__((ext_vector_type(8)))  float  v8f;

__device__ __forceinline__ void atomic_add_f32(float* p, float v) {
    // relaxed, device-scope -> native global_atomic_add_f32 (no CAS loop)
    __hip_atomic_fetch_add(p, v, __ATOMIC_RELAXED, __HIP_MEMORY_SCOPE_AGENT);
}

// ---------------------------------------------------------------------------
// Degree / normalization
// ---------------------------------------------------------------------------
__global__ void init_deg_kernel(float* __restrict__ deg, int n) {
    int i = blockIdx.x * blockDim.x + threadIdx.x;
    if (i < n) deg[i] = 1.0f;           // self-loop contributes 1 to every node
}

__global__ void deg_kernel(const long long* __restrict__ dst,
                           float* __restrict__ deg, long long E) {
    long long e = (long long)blockIdx.x * blockDim.x + threadIdx.x;
    if (e >= E) return;
    atomic_add_f32(&deg[dst[e]], 1.0f);
}

__global__ void rsqrt_kernel(float* __restrict__ dinv, int n) {
    int i = blockIdx.x * blockDim.x + threadIdx.x;
    if (i < n) dinv[i] = rsqrtf(dinv[i]);   // deg >= 1 always (self loop)
}

// ---------------------------------------------------------------------------
// Pre-pack W[Fin,Fout] (f32, row-major) into bf16 laid out exactly as the
// WMMA B operand register file: for each (ktile, coltile), lane L owns 32
// contiguous bytes = 16 bf16 elements with
//   column n = L&15,  K = ktile*32 + (L>>4)*8 + (i&7) + (i>>3)*16   (ISA 7.12.2)
// so the GEMM hot loop fetches a whole B tile with two b128 loads, no cvt.
// ---------------------------------------------------------------------------
template <int NCT>   // NCT = Fout/16
__global__ void pack_w_kernel(const float* __restrict__ W,
                              __bf16* __restrict__ Wp, int Fin) {
    constexpr int Fout = NCT * 16;
    int idx = blockIdx.x * blockDim.x + threadIdx.x;
    int total = Fin * Fout;              // (Fin/32) * NCT * 32 lanes * 16 elems
    if (idx >= total) return;
    int i     = idx & 15;
    int lane  = (idx >> 4) & 31;
    int t     = (idx >> 9) % NCT;
    int ktile = (idx >> 9) / NCT;
    int k   = ktile * 32 + ((lane >> 4) << 3) + (i & 7) + ((i >> 3) << 4);
    int col = t * 16 + (lane & 15);
    Wp[idx] = (__bf16)W[(size_t)k * Fout + col];
}

// ---------------------------------------------------------------------------
// GEMM: H[M,Fout] = X[M,Fin] @ W[Fin,Fout] using v_wmma_f32_16x16x32_bf16.
// One wave computes a 16-row strip, holding all Fout/16 accumulator tiles so
// each X element is loaded (b128-vectorized) and converted to bf16 once.
// B comes pre-packed in register layout: 2x b128 per tile, zero converts.
// ---------------------------------------------------------------------------
template <int NCT>   // NCT = Fout / 16
__global__ __launch_bounds__(256)
void gcn_gemm_wmma(const float* __restrict__ X, const __bf16* __restrict__ Wp,
                   float* __restrict__ H, int M, int Fin) {
    constexpr int Fout = NCT * 16;
    const int wave    = blockIdx.x * (blockDim.x >> 5) + (threadIdx.x >> 5);
    const int nstrips = (M + 15) >> 4;
    if (wave >= nstrips) return;            // wave-uniform: EXEC stays all-ones

    const int lane = threadIdx.x & 31;
    const int m    = lane & 15;             // row within strip / column for C
    const int hs   = (lane >> 4) << 3;      // A: K half-offset (0 or 8)
    const int rowBase = wave << 4;

    const v16bf* Wt = (const v16bf*)Wp;     // [ (ktile*NCT + t)*32 + lane ]

    v8f acc[NCT];
#pragma unroll
    for (int t = 0; t < NCT; ++t) acc[t] = (v8f){0.f,0.f,0.f,0.f,0.f,0.f,0.f,0.f};

    const float* xrow = X + (size_t)(rowBase + m) * Fin;
    for (int k0 = 0; k0 < Fin; k0 += 32) {
        // A tile 16x32 bf16: lane holds row m; two contiguous 8-float runs at
        // K = k0+hs and K = k0+16+hs (ISA 7.12.2). Explicit float4 -> b128.
        const float4 r0 = *(const float4*)(xrow + k0 + hs);
        const float4 r1 = *(const float4*)(xrow + k0 + hs + 4);
        const float4 r2 = *(const float4*)(xrow + k0 + 16 + hs);
        const float4 r3 = *(const float4*)(xrow + k0 + 16 + hs + 4);
        v16bf a;
        a[0]  = (__bf16)r0.x; a[1]  = (__bf16)r0.y; a[2]  = (__bf16)r0.z; a[3]  = (__bf16)r0.w;
        a[4]  = (__bf16)r1.x; a[5]  = (__bf16)r1.y; a[6]  = (__bf16)r1.z; a[7]  = (__bf16)r1.w;
        a[8]  = (__bf16)r2.x; a[9]  = (__bf16)r2.y; a[10] = (__bf16)r2.z; a[11] = (__bf16)r2.w;
        a[12] = (__bf16)r3.x; a[13] = (__bf16)r3.y; a[14] = (__bf16)r3.z; a[15] = (__bf16)r3.w;

        const size_t kbase = (size_t)(k0 >> 5) * NCT;
#pragma unroll
        for (int t = 0; t < NCT; ++t) {
            v16bf b = Wt[(kbase + t) * 32 + lane];      // 32B/lane: 2x b128
            acc[t] = __builtin_amdgcn_wmma_f32_16x16x32_bf16(
                false, a, false, b, (short)0, acc[t], false, false);
        }
    }

    // C/D f32 16x16: VGPR r holds M = r + (lane>>4)*8, N = lane&15.
    const int mo = (lane >> 4) << 3;
#pragma unroll
    for (int t = 0; t < NCT; ++t) {
        float* cp = H + (size_t)rowBase * Fout + t * 16 + m;
#pragma unroll
        for (int r = 0; r < 8; ++r)
            cp[(size_t)(mo + r) * Fout] = acc[t][r];
    }
}

// ---------------------------------------------------------------------------
// Edge scatter: out[dst] += h[src] * dinv[src]*dinv[dst]
// F/4 lanes per edge, float4 gather from L2-resident h, 4x f32 atomics.
// ---------------------------------------------------------------------------
template <int F>
__global__ __launch_bounds__(256)
void scatter_kernel(const long long* __restrict__ src,
                    const long long* __restrict__ dst,
                    const float* __restrict__ h,
                    const float* __restrict__ dinv,
                    float* __restrict__ out, long long E) {
    constexpr int CPE = F / 4;
    long long idx = (long long)blockIdx.x * blockDim.x + threadIdx.x;
    long long e   = idx / CPE;
    int       c   = (int)(idx % CPE);
    if (e >= E) return;
    // stream the edge list ahead of us (global_prefetch_b8)
    __builtin_prefetch(src + e + 4096, 0, 0);
    __builtin_prefetch(dst + e + 4096, 0, 0);

    const long long s = src[e];
    const long long d = dst[e];
    const float     w = dinv[s] * dinv[d];
    const float4 hv = *(const float4*)(h + s * (long long)F + c * 4);
    float* op = out + d * (long long)F + c * 4;
    atomic_add_f32(op + 0, hv.x * w);
    atomic_add_f32(op + 1, hv.y * w);
    atomic_add_f32(op + 2, hv.z * w);
    atomic_add_f32(op + 3, hv.w * w);
}

// ---------------------------------------------------------------------------
// Finish: out = act(out + h * dinv^2 (self-loop) + bias)
// ---------------------------------------------------------------------------
template <bool RELU, int F>
__global__ void finish_kernel(float* __restrict__ out,
                              const float* __restrict__ h,
                              const float* __restrict__ dinv,
                              const float* __restrict__ bias, int n) {
    int idx = blockIdx.x * blockDim.x + threadIdx.x;
    if (idx >= n * F) return;
    int i = idx / F, f = idx % F;
    float d = dinv[i];
    float v = out[idx] + h[idx] * d * d + bias[f];
    if (RELU) v = fmaxf(v, 0.0f);
    out[idx] = v;
}

// ---------------------------------------------------------------------------
extern "C" void kernel_launch(void* const* d_in, const int* in_sizes, int n_in,
                              void* d_out, int out_size, void* d_ws, size_t ws_size,
                              hipStream_t stream) {
    const float*     x  = (const float*)d_in[0];
    const long long* ei = (const long long*)d_in[1];
    const float* W1 = (const float*)d_in[2];
    const float* b1 = (const float*)d_in[3];
    const float* W2 = (const float*)d_in[4];
    const float* b2 = (const float*)d_in[5];
    const float* W3 = (const float*)d_in[6];
    const float* b3 = (const float*)d_in[7];

    const int       N = in_sizes[0] / FEAT;       // 100000
    const long long E = (long long)in_sizes[1] / 2;
    const long long* src = ei;
    const long long* dst = ei + E;

    // Workspace layout: bufH | bufO | dinv | Wp1 | Wp2 | Wp3  (~103 MB)
    float*  bufH = (float*)d_ws;
    float*  bufO = bufH + (size_t)N * HID;
    float*  dinv = bufO + (size_t)N * HID;
    __bf16* Wp1  = (__bf16*)(dinv + N);            // 32B-aligned (prefix % 32 == 0)
    __bf16* Wp2  = Wp1 + FEAT * HID;
    __bf16* Wp3  = Wp2 + HID * HID;

    const int B = 256;
    const int nstrips = (N + 15) / 16;
    const int gemm_blocks = (nstrips + 7) / 8;            // 8 waves per block

    // --- pack weights into WMMA-B bf16 register layout (once per call) ---
    pack_w_kernel<HID / 16><<<(FEAT * HID + B - 1) / B, B, 0, stream>>>(W1, Wp1, FEAT);
    pack_w_kernel<HID / 16><<<(HID * HID + B - 1) / B, B, 0, stream>>>(W2, Wp2, HID);
    pack_w_kernel<OUTF / 16><<<(HID * OUTF + B - 1) / B, B, 0, stream>>>(W3, Wp3, HID);

    // --- normalization: dinv = rsqrt(deg) ---
    init_deg_kernel<<<(N + B - 1) / B, B, 0, stream>>>(dinv, N);
    deg_kernel<<<(int)((E + B - 1) / B), B, 0, stream>>>(dst, dinv, E);
    rsqrt_kernel<<<(N + B - 1) / B, B, 0, stream>>>(dinv, N);

    // --- layer 1: x[N,256] @ W1 -> bufH[N,128]; agg -> bufO; relu ---
    gcn_gemm_wmma<HID / 16><<<gemm_blocks, B, 0, stream>>>(x, Wp1, bufH, N, FEAT);
    hipMemsetAsync(bufO, 0, (size_t)N * HID * sizeof(float), stream);
    {
        long long work = E * (HID / 4);
        scatter_kernel<HID><<<(int)((work + B - 1) / B), B, 0, stream>>>(
            src, dst, bufH, dinv, bufO, E);
    }
    finish_kernel<true, HID><<<((size_t)N * HID + B - 1) / B, B, 0, stream>>>(
        bufO, bufH, dinv, b1, N);

    // --- layer 2: bufO[N,128] @ W2 -> bufH; agg -> bufO (reused); relu ---
    gcn_gemm_wmma<HID / 16><<<gemm_blocks, B, 0, stream>>>(bufO, Wp2, bufH, N, HID);
    hipMemsetAsync(bufO, 0, (size_t)N * HID * sizeof(float), stream);
    {
        long long work = E * (HID / 4);
        scatter_kernel<HID><<<(int)((work + B - 1) / B), B, 0, stream>>>(
            src, dst, bufH, dinv, bufO, E);
    }
    finish_kernel<true, HID><<<((size_t)N * HID + B - 1) / B, B, 0, stream>>>(
        bufO, bufH, dinv, b2, N);

    // --- layer 3: bufO[N,128] @ W3 -> bufH[N,64]; agg -> d_out; no relu ---
    float* out = (float*)d_out;
    gcn_gemm_wmma<OUTF / 16><<<gemm_blocks, B, 0, stream>>>(bufO, Wp3, bufH, N, HID);
    hipMemsetAsync(out, 0, (size_t)N * OUTF * sizeof(float), stream);
    {
        long long work = E * (OUTF / 4);
        scatter_kernel<OUTF><<<(int)((work + B - 1) / B), B, 0, stream>>>(
            src, dst, bufH, dinv, out, E);
    }
    finish_kernel<false, OUTF><<<((size_t)N * OUTF + B - 1) / B, B, 0, stream>>>(
        out, bufH, dinv, b3, N);
}
